// Multi_HeadAttention_26989574488752
// MI455X (gfx1250) — matrix operand order repaired and checked by reading, not compiled, not run on hardware
//
#include <hip/hip_runtime.h>

// ---------------------------------------------------------------------------
// MHA forward for MI455X (gfx1250, wave32, WMMA bf16 16x16x32, f32 accum)
// B=2, S=2048, D=1024, H=16, DK=DV=64
// Pipeline: f32->bf16 convert; QKV WMMA GEMMs (LDS-staged B tile via
// GLOBAL_LOAD_ASYNC_TO_LDS_B128, double-buffered, raw split barriers,
// peeled epilogue); flash-attention (WMMA + online softmax, prefetched
// K/V fragments); final WMMA GEMM -> f32.
// ---------------------------------------------------------------------------

typedef __attribute__((ext_vector_type(16))) __bf16 v16bf;
typedef __attribute__((ext_vector_type(8)))  float  v8f;

constexpr int cB = 2, cS = 2048, cD = 1024, cH = 16;

#define USE_ASYNC_COPY 1

__device__ __forceinline__ unsigned short f2bf(float f) {
  unsigned int x = __float_as_uint(f);
  unsigned int r = (x + 0x7FFFu + ((x >> 16) & 1u)) >> 16;   // RNE
  return (unsigned short)r;
}

union Frag16 { v16bf v; uint4 q[2]; };

// A fragment (16 x K=32, bf16), source row-major, leading dim ld.
__device__ __forceinline__ v16bf load_a_frag(const unsigned short* base, int ld,
                                             int kb, int lane) {
  const int row = lane & 15;
  const int hi8 = (lane >> 4) << 3;            // 0 or 8
  const unsigned short* p = base + (size_t)row * ld + kb + hi8;
  Frag16 f;
  f.q[0] = *reinterpret_cast<const uint4*>(p);
  f.q[1] = *reinterpret_cast<const uint4*>(p + 16);
  return f.v;
}

// B fragment (K=32 x 16, bf16). Source row-major [N,K]: B(k,n)=Brow[n,k].
__device__ __forceinline__ v16bf load_b_frag(const unsigned short* base, int ld,
                                             int kb, int lane) {
  const int col  = lane & 15;
  const int hi8  = (lane >> 4) << 3;           // 0 or 8
  const unsigned short* p = base + (size_t)col * ld + kb + hi8;
  Frag16 f;
  f.q[0] = *reinterpret_cast<const uint4*>(p);
  f.q[1] = *reinterpret_cast<const uint4*>(p + 16);
  return f.v;
}

__device__ __forceinline__ v8f wmma_bf16(v16bf a, v16bf b, v8f c) {
  return __builtin_amdgcn_wmma_f32_16x16x32_bf16(false, a, false, b,
                                                 (short)0, c, false, false);
}

// 16-byte global -> LDS copy via the gfx1250 async engine (ASYNCcnt-tracked).
__device__ __forceinline__ void copy16_g2l(const unsigned short* gsrc,
                                           unsigned short* ldst) {
#if USE_ASYNC_COPY
  unsigned loff = (unsigned)(size_t)ldst;      // LDS aperture: low 32b = offset
  asm volatile("global_load_async_to_lds_b128 %0, %1, off"
               :: "v"(loff), "v"(gsrc) : "memory");
#else
  *reinterpret_cast<uint4*>(ldst) = *reinterpret_cast<const uint4*>(gsrc);
#endif
}

__device__ __forceinline__ void async_wait_le1() {
#if USE_ASYNC_COPY
  asm volatile("s_wait_asynccnt 1" ::: "memory");
#endif
}
__device__ __forceinline__ void async_wait_0() {
#if USE_ASYNC_COPY
  asm volatile("s_wait_asynccnt 0" ::: "memory");
#else
  __builtin_amdgcn_s_waitcnt(0);  // no-op placeholder; sync path needs none
#endif
}

// Raw workgroup split barrier: unlike __syncthreads() this does NOT drain
// LOADcnt (global loads to VGPRs need no cross-wave visibility). All LDS
// producers (async copies) are explicitly fenced with s_wait_asynccnt before
// the signal, and every ds_load is dscnt-waited by its consuming WMMA before
// the wave reaches the next signal.
__device__ __forceinline__ void wg_barrier() {
  asm volatile("s_barrier_signal -1\n\ts_barrier_wait -1" ::: "memory");
}

// ---------------------------------------------------------------------------
// f32 -> bf16 conversion
// ---------------------------------------------------------------------------
__global__ void f32_to_bf16_kernel(const float* __restrict__ src,
                                   unsigned short* __restrict__ dst, int n) {
  int i = blockIdx.x * blockDim.x + threadIdx.x;
  if (i < n) dst[i] = f2bf(src[i]);
}

// ---------------------------------------------------------------------------
// GEMM: out[m,n] = sum_k A[m,k] * W[n,k] + bias[n]
// A bf16 [M,K] rm ; W bf16 [N,K] rm. Block 256 thr (8 waves): 128x64 tile,
// wave tile 16x64. B tile (64x32) in LDS, double-buffered async copy;
// A fragment register-prefetched; last K-step peeled -> branch-free loop.
// mode 0: Q -> [B,H,S,64] bf16 * 0.125 ; mode 1: K -> [B,H,S,64] bf16
// mode 2: V^T -> [B,H,64,S] bf16       ; mode 3: f32 -> out_f32[m*N+n]
// ---------------------------------------------------------------------------
__global__ __launch_bounds__(256) void gemm_bf16_kernel(
    const unsigned short* __restrict__ A, const unsigned short* __restrict__ W,
    const float* __restrict__ bias, unsigned short* __restrict__ out_bf,
    float* __restrict__ out_f32, int M, int N, int K, int mode) {
  __shared__ __align__(16) unsigned short Bsh[2][64 * 32];   // 2 x 4 KB

  const int tid = threadIdx.x;
  const int ln = tid & 31;
  const int w  = tid >> 5;
  const int m0 = blockIdx.y * 128 + w * 16;
  const int n0 = blockIdx.x * 64;
  const int lc = ln & 15;
  const int hi = ln >> 4;

  // Staging map: thread t copies 16B: n = t/4, k = (t%4)*8
  const int sn = tid >> 2;
  const int sk = (tid & 3) << 3;
  const unsigned short* Wsrc = W + (size_t)(n0 + sn) * K + sk;
  const int ldsoff = sn * 32 + sk;

  const v8f vzero = {0.f, 0.f, 0.f, 0.f, 0.f, 0.f, 0.f, 0.f};
  v8f acc[4] = {vzero, vzero, vzero, vzero};

  const unsigned short* Abase = A + (size_t)m0 * K;
  const int nsteps = K / 32;                     // >= 2 for all our shapes

  copy16_g2l(Wsrc, &Bsh[0][ldsoff]);             // stage k-step 0
  v16bf a_cur = load_a_frag(Abase, K, 0, ln);

  // Steady state: stages step i+1 while computing step i. Unroll by 2 so the
  // A-fragment double buffer ping-pongs registers instead of copying.
#pragma unroll 2
  for (int i = 0; i < nsteps - 1; ++i) {
    const int buf = i & 1;
    wg_barrier();                                // buf^1 consumed by all waves
    copy16_g2l(Wsrc + (i + 1) * 32, &Bsh[buf ^ 1][ldsoff]);
    v16bf a_nxt = load_a_frag(Abase, K, (i + 1) * 32, ln);
    async_wait_le1();                            // step i's copies landed
    wg_barrier();                                // ... and are visible
    v16bf bf[4];
#pragma unroll
    for (int j = 0; j < 4; ++j)
      bf[j] = load_b_frag(&Bsh[buf][j * 16 * 32], 32, 0, ln);
#pragma unroll
    for (int j = 0; j < 4; ++j) acc[j] = wmma_bf16(a_cur, bf[j], acc[j]);
    a_cur = a_nxt;
  }

  // Peeled last step.
  {
    const int buf = (nsteps - 1) & 1;
    async_wait_0();
    wg_barrier();
    v16bf bf[4];
#pragma unroll
    for (int j = 0; j < 4; ++j)
      bf[j] = load_b_frag(&Bsh[buf][j * 16 * 32], 32, 0, ln);
#pragma unroll
    for (int j = 0; j < 4; ++j) acc[j] = wmma_bf16(a_cur, bf[j], acc[j]);
  }

#pragma unroll
  for (int j = 0; j < 4; ++j) {
    const int n  = n0 + j * 16 + lc;
    const float bj = bias[n];
#pragma unroll
    for (int r = 0; r < 8; ++r) {
      const int m = m0 + r + hi * 8;
      float val = acc[j][r] + bj;
      if (mode == 0 || mode == 1) {
        if (mode == 0) val *= 0.125f;
        const int bb = m / cS, s = m % cS, hh = n >> 6, d = n & 63;
        out_bf[(((size_t)(bb * cH + hh) * cS + s) << 6) + d] = f2bf(val);
      } else if (mode == 2) {
        const int bb = m / cS, s = m % cS, hh = n >> 6, d = n & 63;
        out_bf[((size_t)((bb * cH + hh) << 6) + d) * cS + s] = f2bf(val);
      } else {
        out_f32[(size_t)m * N + n] = val;
      }
    }
  }
}

// ---------------------------------------------------------------------------
// Flash attention: one wave per 16-query tile of one (b,h).
// Q: [B,H,S,64] bf16 (pre-scaled), K: [B,H,S,64] bf16, Vt: [B,H,64,S] bf16.
// O: [B,S,H*64] bf16. K fragments prefetched one chunk ahead; V fragments
// issued before the softmax VALU section so their wait is fully hidden.
// ---------------------------------------------------------------------------
__global__ __launch_bounds__(128) void attn_kernel(
    const unsigned short* __restrict__ Qb, const unsigned short* __restrict__ Kb,
    const unsigned short* __restrict__ Vt, const float* __restrict__ pmask,
    unsigned short* __restrict__ Ob) {
  __shared__ __align__(16) unsigned short plds[4][16 * 32];  // per-wave P tile

  const int ln = threadIdx.x & 31;
  const int w  = threadIdx.x >> 5;
  const int gw = blockIdx.x * 4 + w;
  const int qt = gw % (cS / 16);
  const int t  = gw / (cS / 16);
  const int h  = t % cH;
  const int b  = t / cH;
  const int qb = qt * 16;
  const int lc = ln & 15;
  const int hi = ln >> 4;

  const unsigned short* Qp = Qb + ((size_t)(b * cH + h) * cS + qb) * 64;
  const unsigned short* Kp = Kb + (size_t)(b * cH + h) * cS * 64;
  const unsigned short* Vp = Vt + (size_t)(b * cH + h) * 64 * cS;
  const float* pm = pmask + (size_t)b * cS;

  const v16bf qa0 = load_a_frag(Qp, 64, 0, ln);
  const v16bf qa1 = load_a_frag(Qp, 64, 32, ln);

  const v8f vzero = {0.f, 0.f, 0.f, 0.f, 0.f, 0.f, 0.f, 0.f};
  v8f o[4] = {vzero, vzero, vzero, vzero};
  float rmax[8], rsum[8];
#pragma unroll
  for (int r = 0; r < 8; ++r) { rmax[r] = -1e30f; rsum[r] = 0.f; }

  unsigned short* lp = plds[w];
  const int kend = qb + 16;                       // causal: keys in [0, qb+16)

  // Prefetch K fragments for the first chunk: kbf[nt][kk]
  v16bf kbf[2][2];
#pragma unroll
  for (int nt = 0; nt < 2; ++nt)
#pragma unroll
    for (int kk = 0; kk < 2; ++kk)
      kbf[nt][kk] = load_b_frag(Kp + (size_t)(nt * 16) * 64, 64, kk * 32, ln);

  for (int kc = 0; kc < kend; kc += 32) {
    // ---- issue V fragment loads now (consumed after softmax) ----
    v16bf vb[4];
#pragma unroll
    for (int j = 0; j < 4; ++j)
      vb[j] = load_b_frag(Vp + (size_t)(j * 16) * cS, cS, kc, ln);

    // ---- S = Q K^T from prefetched K fragments ----
    v8f st[2];
#pragma unroll
    for (int nt = 0; nt < 2; ++nt) {
      v8f s = vzero;
      s = wmma_bf16(qa0, kbf[nt][0], s);
      s = wmma_bf16(qa1, kbf[nt][1], s);
      st[nt] = s;
    }

    // ---- prefetch K fragments for next chunk ----
    if (kc + 32 < kend) {
#pragma unroll
      for (int nt = 0; nt < 2; ++nt)
#pragma unroll
        for (int kk = 0; kk < 2; ++kk)
          kbf[nt][kk] = load_b_frag(Kp + (size_t)(kc + 32 + nt * 16) * 64, 64,
                                    kk * 32, ln);
    }

    // ---- mask + online softmax ----
    float sv[2][8];
#pragma unroll
    for (int nt = 0; nt < 2; ++nt) {
      const int key = kc + nt * 16 + lc;
      const float pn = (pm[key] > 0.f) ? -__builtin_inff() : 0.f;
#pragma unroll
      for (int r = 0; r < 8; ++r) {
        const int q = qb + r + hi * 8;
        float v = st[nt][r] + pn;
        if (key > q) v = -__builtin_inff();
        sv[nt][r] = v;
      }
    }

    float scale[8];
#pragma unroll
    for (int r = 0; r < 8; ++r) {
      float mx = fmaxf(sv[0][r], sv[1][r]);
#pragma unroll
      for (int off = 8; off >= 1; off >>= 1) mx = fmaxf(mx, __shfl_xor(mx, off));
      const float nm = fmaxf(rmax[r], mx);        // >= -1e30, always finite
      scale[r] = __expf(rmax[r] - nm);
      rmax[r] = nm;
      const float p0 = __expf(sv[0][r] - nm);
      const float p1 = __expf(sv[1][r] - nm);
      sv[0][r] = p0; sv[1][r] = p1;
      float sm = p0 + p1;
#pragma unroll
      for (int off = 8; off >= 1; off >>= 1) sm += __shfl_xor(sm, off);
      rsum[r] = rsum[r] * scale[r] + sm;
    }

    // rescale running O
#pragma unroll
    for (int j = 0; j < 4; ++j)
#pragma unroll
      for (int r = 0; r < 8; ++r) o[j][r] *= scale[r];

    // ---- P (C-layout) -> LDS row-major -> A-fragment layout ----
#pragma unroll
    for (int nt = 0; nt < 2; ++nt)
#pragma unroll
      for (int r = 0; r < 8; ++r)
        lp[(r + hi * 8) * 32 + nt * 16 + lc] = f2bf(sv[nt][r]);
    v16bf pa = load_a_frag(lp, 32, 0, ln);

    // ---- O += P * V ----
#pragma unroll
    for (int j = 0; j < 4; ++j) o[j] = wmma_bf16(pa, vb[j], o[j]);
  }

  // ---- normalize and store O[b, q, h*64 + dv] as bf16 ----
  float inv[8];
#pragma unroll
  for (int r = 0; r < 8; ++r) inv[r] = 1.0f / rsum[r];
#pragma unroll
  for (int j = 0; j < 4; ++j)
#pragma unroll
    for (int r = 0; r < 8; ++r) {
      const int q = qb + r + hi * 8;
      Ob[((size_t)(b * cS + q) << 10) + h * 64 + j * 16 + lc] =
          f2bf(o[j][r] * inv[r]);
    }
}

// ---------------------------------------------------------------------------
// Host-side orchestration
// ---------------------------------------------------------------------------
extern "C" void kernel_launch(void* const* d_in, const int* in_sizes, int n_in,
                              void* d_out, int out_size, void* d_ws, size_t ws_size,
                              hipStream_t stream) {
  const float* input = (const float*)d_in[0];
  const float* pmask = (const float*)d_in[1];
  const float* wq_w  = (const float*)d_in[2];
  const float* wq_b  = (const float*)d_in[3];
  const float* wk_w  = (const float*)d_in[4];
  const float* wk_b  = (const float*)d_in[5];
  const float* wv_w  = (const float*)d_in[6];
  const float* wv_b  = (const float*)d_in[7];
  const float* fin_w = (const float*)d_in[8];
  const float* fin_b = (const float*)d_in[9];
  float* out = (float*)d_out;

  const size_t MTOK = (size_t)cB * cS;           // 4096 tokens
  const size_t NIN  = MTOK * cD;                 // 4 Mi elements
  const size_t NW   = (size_t)cD * cD;           // 1 Mi elements per weight

  unsigned short* ws    = (unsigned short*)d_ws;
  unsigned short* in_bf = ws;
  unsigned short* wqbf  = in_bf + NIN;
  unsigned short* wkbf  = wqbf + NW;
  unsigned short* wvbf  = wkbf + NW;
  unsigned short* wfbf  = wvbf + NW;
  unsigned short* q_bf  = wfbf + NW;             // [B,H,S,64]
  unsigned short* k_bf  = q_bf + NIN;            // [B,H,S,64]
  unsigned short* vt_bf = k_bf + NIN;            // [B,H,64,S]
  unsigned short* o_bf  = vt_bf + NIN;           // [B,S,H*64]
  // total: 48 MiB of d_ws

  const int CT = 256;
  f32_to_bf16_kernel<<<dim3((NIN + CT - 1) / CT), CT, 0, stream>>>(input, in_bf, (int)NIN);
  f32_to_bf16_kernel<<<dim3((NW + CT - 1) / CT), CT, 0, stream>>>(wq_w, wqbf, (int)NW);
  f32_to_bf16_kernel<<<dim3((NW + CT - 1) / CT), CT, 0, stream>>>(wk_w, wkbf, (int)NW);
  f32_to_bf16_kernel<<<dim3((NW + CT - 1) / CT), CT, 0, stream>>>(wv_w, wvbf, (int)NW);
  f32_to_bf16_kernel<<<dim3((NW + CT - 1) / CT), CT, 0, stream>>>(fin_w, wfbf, (int)NW);

  const dim3 ggrid(cD / 64, (unsigned)(MTOK / 128));   // (16, 32)
  gemm_bf16_kernel<<<ggrid, 256, 0, stream>>>(in_bf, wqbf, wq_b, q_bf, nullptr,
                                              (int)MTOK, cD, cD, 0);
  gemm_bf16_kernel<<<ggrid, 256, 0, stream>>>(in_bf, wkbf, wk_b, k_bf, nullptr,
                                              (int)MTOK, cD, cD, 1);
  gemm_bf16_kernel<<<ggrid, 256, 0, stream>>>(in_bf, wvbf, wv_b, vt_bf, nullptr,
                                              (int)MTOK, cD, cD, 2);

  const int nwaves = cB * cH * (cS / 16);        // 4096 waves
  attn_kernel<<<dim3(nwaves / 4), 128, 0, stream>>>(q_bf, k_bf, vt_bf, pmask, o_bf);

  gemm_bf16_kernel<<<ggrid, 256, 0, stream>>>(o_bf, wfbf, fin_b, nullptr, out,
                                              (int)MTOK, cD, cD, 3);
}
